// Conv3DSubmModule_8598524527254
// MI455X (gfx1250) — matrix operand order repaired and checked by reading, not compiled, not run on hardware
//
#include <hip/hip_runtime.h>

// Submanifold sparse 3D conv as implicit GEMM on CDNA5 (gfx1250) WMMA.
//
//   out[n, :] = sum_k feats[idx[n,k], :] @ W[k]     (idx < 0 => skip)
//
// N=100000, K3=27, D=DP=64, fp32. 22.1 GFLOP; feats (25.6MB) + weight
// (0.44MB) are L2-resident in MI455X's 192MB L2 -> compute-bound.
//
// Split-f16 3-product scheme on V_WMMA_F32_16X16X32_F16 (16384 FLOP/instr
// vs 2048 for the fp32-K4 pipe):
//     x = hi + lo (f16 each, ~22-bit significand),
//     A*B ~= Ahi*Bhi + Ahi*Blo + Alo*Bhi   (fp32 accumulate, err ~2^-22)
// => 48 f16 WMMAs per tap per wave, gather bytes identical to fp32.
// Weights pre-swizzled into the exact per-lane B-register layout: staging
// is a raw 16KB copy, every B fragment a contiguous 32B ds read per lane.
//
// Round-4 change: product-major WMMA ordering per chunk (all 8 B fragments
// resident) -> same-accumulator distance 8 (> the 5-deep f16 WMMA hazard
// window), removing the s_delay_alu TRANS32 stalls seen in round 3.
// Falls back to the f32 WMMA kernel if d_ws is too small (~26.1MB needed).

typedef float     v2f  __attribute__((ext_vector_type(2)));
typedef float     v8f  __attribute__((ext_vector_type(8)));
typedef _Float16  v8h  __attribute__((ext_vector_type(8)));
typedef _Float16  v16h __attribute__((ext_vector_type(16)));

#define D_IN   64
#define D_OUT  64
#define KVOL   27

// Swizzled weight layout: [k][comp(hi,lo)][chunk(2)][col(4)][lane(32)][16h]
// B-register layout (16x16x32 f16): lane L: N = L&15, halves h=0..15 map to
// K = (L>>4)*8 + (h&7) + (h>>3)*16 within the 32-deep chunk.
#define WTAP_HALVES (2 * 2 * 4 * 32 * 16)     // 8192 halves = 16 KB per tap

__global__ __launch_bounds__(256) void prep_split_feats(
    const float* __restrict__ feats, _Float16* __restrict__ featsH, int total)
{
    for (int e = blockIdx.x * 256 + threadIdx.x; e < total;
         e += gridDim.x * 256) {
        const float x  = feats[e];
        const _Float16 hi = (_Float16)x;
        featsH[e]         = hi;
        featsH[total + e] = (_Float16)(x - (float)hi);
    }
}

__global__ __launch_bounds__(256) void prep_swizzle_weight(
    const float* __restrict__ weight, _Float16* __restrict__ swzW)
{
    const int flat = blockIdx.x * 256 + threadIdx.x;   // [0, 27*2*4096)
    if (flat >= KVOL * 2 * 4096) return;
    const int h    = flat & 15;
    const int lane = (flat >> 4) & 31;
    const int col  = (flat >> 9) & 3;
    const int chunk= (flat >> 11) & 1;
    const int comp = (flat >> 12) & 1;
    const int k    = flat >> 13;
    const int d = chunk * 32 + (lane >> 4) * 8 + (h & 7) + (h >> 3) * 16;
    const int n = col * 16 + (lane & 15);
    const float x = weight[(k * D_IN + d) * D_OUT + n];
    const _Float16 hi = (_Float16)x;
    swzW[flat] = (comp == 0) ? hi : (_Float16)(x - (float)hi);
}

#define WMMA_F16(A, B, C) \
    __builtin_amdgcn_wmma_f32_16x16x32_f16(false, (A), false, (B), (short)0, (C), false, false)

// Block: 256 threads = 8 wave32. Wave owns 32 voxel rows x 64 out channels.
__global__ __launch_bounds__(256) void subm_conv3d_wmma_f16split(
    const _Float16* __restrict__ featsH,   // [2][N][64] hi plane, lo plane
    const int*      __restrict__ indices,  // [N][27]
    const _Float16* __restrict__ swzW,     // [27][WTAP_HALVES] pre-swizzled
    float*          __restrict__ out,      // [N][64]
    int n_voxels)
{
    __shared__ _Float16 lds_w[2][WTAP_HALVES];   // 32 KB double buffer

    const int tid  = threadIdx.x;
    const int wave = tid >> 5;
    const int lane = tid & 31;
    const int half = lane >> 4;
    const int l16  = lane & 15;
    const int feats_plane = n_voxels * D_IN;

    const int wave_base = blockIdx.x * 256 + wave * 32;

    v8f acc[2][4] = {};

    // stage tap 0 (straight copy: pre-swizzled)
    {
        const uint4* src = (const uint4*)swzW;          // 1024 uint4 per tap
        uint4* dst = (uint4*)lds_w[0];
        #pragma unroll
        for (int i = 0; i < 4; ++i) dst[tid + i * 256] = src[tid + i * 256];
    }
    __syncthreads();

    for (int k = 0; k < KVOL; ++k) {
        // prefetch next tap's swizzled weights into registers
        uint4 wreg[4];
        const bool have_next = (k + 1) < KVOL;
        if (have_next) {
            const uint4* src = (const uint4*)(swzW + (k + 1) * WTAP_HALVES);
            #pragma unroll
            for (int i = 0; i < 4; ++i) wreg[i] = src[tid + i * 256];
        }

        const _Float16* cur = lds_w[k & 1];

        // Gather A fragments. 16-bit A layout (16x32): lane half 0 needs
        // K = [0..7] & [16..23]; half 1 needs K = [8..15] & [24..31]
        // (within each 32-deep chunk) -> two contiguous 16B runs per lane.
        v16h afr[2][2][2];   // [row r][comp][chunk]
        #pragma unroll
        for (int r = 0; r < 2; ++r) {
            const int my_row = wave_base + r * 16 + l16;
            const int idxv = (my_row < n_voxels) ? indices[my_row * KVOL + k] : -1;
            const bool valid = idxv >= 0;
            const long long rowoff = (long long)(valid ? idxv : 0) << 6;
            #pragma unroll
            for (int comp = 0; comp < 2; ++comp) {
                const _Float16* src = featsH + comp * feats_plane + rowoff;
                #pragma unroll
                for (int chunk = 0; chunk < 2; ++chunk) {
                    v16h a = {};
                    if (valid) {
                        const int base = chunk * 32 + half * 8;
                        const v8h r0 = *(const v8h*)(src + base);
                        const v8h r1 = *(const v8h*)(src + base + 16);
                        a = __builtin_shufflevector(r0, r1,
                              0,1,2,3,4,5,6,7, 8,9,10,11,12,13,14,15);
                    }
                    afr[r][comp][chunk] = a;
                }
            }
        }

        // 2 chunks x (4 cols x 2 rows x 3 products) = 48 f16 WMMAs per tap.
        // All 8 B fragments of a chunk loaded up front, then product-major
        // sweep: same-accumulator WMMA distance = 8 (no TRANS32 stalls).
        #pragma unroll
        for (int chunk = 0; chunk < 2; ++chunk) {
            v16h bh[4], bl[4];
            #pragma unroll
            for (int col = 0; col < 4; ++col) {
                const int sub_hi = (0 * 2 + chunk) * 4 + col;
                const int sub_lo = (1 * 2 + chunk) * 4 + col;
                bh[col] = *(const v16h*)&cur[(sub_hi * 32 + lane) * 16];
                bl[col] = *(const v16h*)&cur[(sub_lo * 32 + lane) * 16];
            }
            #pragma unroll
            for (int col = 0; col < 4; ++col)
                #pragma unroll
                for (int r = 0; r < 2; ++r)
                    acc[r][col] = WMMA_F16(afr[r][0][chunk], bl[col], acc[r][col]);
            #pragma unroll
            for (int col = 0; col < 4; ++col)
                #pragma unroll
                for (int r = 0; r < 2; ++r)
                    acc[r][col] = WMMA_F16(afr[r][1][chunk], bh[col], acc[r][col]);
            #pragma unroll
            for (int col = 0; col < 4; ++col)
                #pragma unroll
                for (int r = 0; r < 2; ++r)
                    acc[r][col] = WMMA_F16(afr[r][0][chunk], bh[col], acc[r][col]);
        }

        if (have_next) {
            uint4* dst = (uint4*)lds_w[(k + 1) & 1];
            #pragma unroll
            for (int i = 0; i < 4; ++i) dst[tid + i * 256] = wreg[i];
        }
        __syncthreads();
    }

    // C/D layout: VGPR j -> M = j (lanes 0-15) / j+8 (lanes 16-31), N = l16.
    #pragma unroll
    for (int r = 0; r < 2; ++r)
        #pragma unroll
        for (int col = 0; col < 4; ++col)
            #pragma unroll
            for (int j = 0; j < 8; ++j) {
                const int row = wave_base + r * 16 + half * 8 + j;
                if (row < n_voxels)
                    out[(long long)row * D_OUT + col * 16 + l16] = acc[r][col][j];
            }
}

// ---------------- f32 WMMA fallback ----------------
#define PAIR_STRIDE 160
#define WBUF_DWORDS (32 * PAIR_STRIDE)
#define WMMA_F32(A, B, C) \
    __builtin_amdgcn_wmma_f32_16x16x4_f32(false, (A), false, (B), (short)0, (C), false, false)

__global__ __launch_bounds__(256) void subm_conv3d_wmma_f32(
    const float* __restrict__ feats, const int* __restrict__ indices,
    const float* __restrict__ weight, float* __restrict__ out, int n_voxels)
{
    __shared__ float lds_w[2][WBUF_DWORDS];
    const int tid = threadIdx.x, wave = tid >> 5, lane = tid & 31;
    const int half = lane >> 4, l16 = lane & 15;
    const int wave_base = blockIdx.x * 256 + wave * 32;
    v8f acc[2][4] = {};
    {
        const float4* wk = (const float4*)weight;
        #pragma unroll
        for (int i = 0; i < 4; ++i) {
            const int e4 = tid + i * 256;
            const float4 v = wk[e4];
            const int elem = e4 * 4, d = elem >> 6, n = elem & 63;
            float* dst = &lds_w[0][(d >> 1) * PAIR_STRIDE + (d & 1)];
            dst[(n+0)*2] = v.x; dst[(n+1)*2] = v.y;
            dst[(n+2)*2] = v.z; dst[(n+3)*2] = v.w;
        }
    }
    __syncthreads();
    for (int k = 0; k < KVOL; ++k) {
        float4 wreg[4];
        const bool have_next = (k + 1) < KVOL;
        if (have_next) {
            const float4* wk = (const float4*)(weight + (k + 1) * (D_IN * D_OUT));
            #pragma unroll
            for (int i = 0; i < 4; ++i) wreg[i] = wk[tid + i * 256];
        }
        const float* cur = lds_w[k & 1];
        v2f afrag[2][16];
        #pragma unroll
        for (int r = 0; r < 2; ++r) {
            const int my_row = wave_base + r * 16 + l16;
            const int idxv = (my_row < n_voxels) ? indices[my_row * KVOL + k] : -1;
            const bool valid = idxv >= 0;
            const float* src = feats + ((long long)(valid ? idxv : 0) << 6);
            #pragma unroll
            for (int dstep = 0; dstep < 16; ++dstep) {
                v2f a = {0.f, 0.f};
                if (valid) {
                    const float2 t = *(const float2*)(src + dstep * 4 + half * 2);
                    a.x = t.x; a.y = t.y;
                }
                afrag[r][dstep] = a;
            }
        }
        #pragma unroll
        for (int dstep = 0; dstep < 16; ++dstep) {
            const float* prow = &cur[(dstep * 2 + half) * PAIR_STRIDE + l16 * 2];
            #pragma unroll
            for (int col = 0; col < 4; ++col) {
                const v2f b = *(const v2f*)(prow + col * 32);
                acc[0][col] = WMMA_F32(afrag[0][dstep], b, acc[0][col]);
                acc[1][col] = WMMA_F32(afrag[1][dstep], b, acc[1][col]);
            }
        }
        if (have_next) {
            float* nxt = lds_w[(k + 1) & 1];
            #pragma unroll
            for (int i = 0; i < 4; ++i) {
                const int e4 = tid + i * 256, elem = e4 * 4;
                const int d = elem >> 6, n = elem & 63;
                float* dst = &nxt[(d >> 1) * PAIR_STRIDE + (d & 1)];
                dst[(n+0)*2] = wreg[i].x; dst[(n+1)*2] = wreg[i].y;
                dst[(n+2)*2] = wreg[i].z; dst[(n+3)*2] = wreg[i].w;
            }
        }
        __syncthreads();
    }
    #pragma unroll
    for (int r = 0; r < 2; ++r)
        #pragma unroll
        for (int col = 0; col < 4; ++col)
            #pragma unroll
            for (int j = 0; j < 8; ++j) {
                const int row = wave_base + r * 16 + half * 8 + j;
                if (row < n_voxels)
                    out[(long long)row * D_OUT + col * 16 + l16] = acc[r][col][j];
            }
}

// ---------------- launcher ----------------
extern "C" void kernel_launch(void* const* d_in, const int* in_sizes, int n_in,
                              void* d_out, int out_size, void* d_ws, size_t ws_size,
                              hipStream_t stream) {
    const float* feats   = (const float*)d_in[0];
    const int*   indices = (const int*)  d_in[1];   // harness: integer -> int32
    const float* weight  = (const float*)d_in[2];
    float*       out     = (float*)d_out;

    const int n_voxels = in_sizes[0] / D_IN;        // N = 100000
    const int blocks   = (n_voxels + 255) / 256;    // 256 voxels per block

    const size_t feats_halves = (size_t)n_voxels * D_IN * 2;      // hi+lo
    const size_t w_halves     = (size_t)KVOL * WTAP_HALVES;
    const size_t ws_needed    = (feats_halves + w_halves) * sizeof(_Float16);

    if (ws_size >= ws_needed) {
        _Float16* featsH = (_Float16*)d_ws;                       // [2][N][64]
        _Float16* swzW   = featsH + feats_halves;                 // [27][8192]

        const int total = n_voxels * D_IN;
        prep_split_feats<<<(total + 2047) / 2048, 256, 0, stream>>>(
            feats, featsH, total);
        prep_swizzle_weight<<<(KVOL * 2 * 4096 + 255) / 256, 256, 0, stream>>>(
            weight, swzW);
        subm_conv3d_wmma_f16split<<<blocks, 256, 0, stream>>>(
            featsH, indices, swzW, out, n_voxels);
    } else {
        subm_conv3d_wmma_f32<<<blocks, 256, 0, stream>>>(
            feats, indices, weight, out, n_voxels);
    }
}
